// GCN_20452634264145
// MI455X (gfx1250) — matrix operand order repaired and checked by reading, not compiled, not run on hardware
//
#include <hip/hip_runtime.h>
#include <hip/hip_bf16.h>
#include <stdint.h>

// ---------------- types for CDNA5 WMMA ----------------
typedef __attribute__((ext_vector_type(16))) __bf16 v16bf;
typedef __attribute__((ext_vector_type(8)))  float  v8f;

union Frag16 { uint4 q[2]; v16bf v; };   // 32 bytes: 16 bf16 elements

#define N_NODES 50000
#define N_EDGES 800000
#define NFEAT   512
#define NHID    256
#define NCLASS  64

__device__ __forceinline__ unsigned short f2bf(float f) {
    uint32_t u = __float_as_uint(f);
    u += 0x7FFFu + ((u >> 16) & 1u);     // round-to-nearest-even
    return (unsigned short)(u >> 16);
}

// ---------------- fp32 -> bf16 conversion ----------------
__global__ __launch_bounds__(256)
void k_f32_to_bf16(const float* __restrict__ in, unsigned short* __restrict__ out, int n) {
    int i = blockIdx.x * 256 + threadIdx.x;
    if (i < n) out[i] = f2bf(in[i]);
}

// fp32 [K,N] -> bf16 [N,K] (pre-transposed weights: kills the in-LDS transpose)
__global__ __launch_bounds__(256)
void k_wtrans_bf16(const float* __restrict__ in, unsigned short* __restrict__ out,
                   int K, int N) {
    int i = blockIdx.x * 256 + threadIdx.x;
    if (i < K * N) {
        int n = i / K, k = i - n * K;
        out[i] = f2bf(in[(size_t)k * N + n]);
    }
}

// ---------------- bf16 WMMA GEMM: C[M,N] = A[M,K] @ Bt[N,K]^T (+bias) ----------------
// Block tile 128x64, 8 waves (4x2), each wave 32x32 = 2x2 WMMA 16x16 tiles.
// K-step 32, double-buffered LDS (ping-pong), register-staged prefetch.
#define BM 128
#define BN 64
#define BK 32

__global__ __launch_bounds__(256)
void k_gemm_bf16(const unsigned short* __restrict__ A,   // [M,K] bf16
                 const unsigned short* __restrict__ Bt,  // [N,K] bf16 (transposed)
                 const float* __restrict__ bias,         // [N] or nullptr
                 float* __restrict__ C,                   // [M,N]
                 int M, int N, int K)
{
    __shared__ unsigned short As[2][BM][BK];   // row-major (k contiguous) 16 KB
    __shared__ unsigned short Bs[2][BN][BK];   // n-major  (k contiguous)  8 KB

    const int tid  = threadIdx.x;
    const int wave = tid >> 5;
    const int lane = tid & 31;
    const int wm   = wave >> 1;          // 0..3
    const int wn   = wave & 1;           // 0..1
    const int bm   = blockIdx.x * BM;
    const int bn   = blockIdx.y * BN;

    const int l15  = lane & 15;
    const int half = lane >> 4;          // 0 or 1
    const int kbase = half * 8;

    v8f c[2][2] = {};

    // A loader: 128 rows x 32 bf16; 2 threads per row, 16 bf16 (2x uint4) each
    const int arow = tid >> 1;
    const int aseg = (tid & 1) * 16;
    // B loader (first 128 threads): 64 n-rows x 32 bf16; 2 threads per row
    const bool bstage = tid < 128;
    const int brow = tid >> 1;           // 0..63 when bstage
    const int bseg = aseg;

    const int gr = bm + arow;
    const unsigned short* aptr = A  + (size_t)gr * K + aseg;
    const unsigned short* bptr = Bt + (size_t)(bn + brow) * K + bseg;

    // ---- prologue: stage tile 0 ----
    uint4 ra0 = {0,0,0,0}, ra1 = {0,0,0,0}, rb0 = {0,0,0,0}, rb1 = {0,0,0,0};
    if (gr < M) { ra0 = ((const uint4*)aptr)[0]; ra1 = ((const uint4*)aptr)[1]; }
    if (bstage) { rb0 = ((const uint4*)bptr)[0]; rb1 = ((const uint4*)bptr)[1]; }
    *(uint4*)&As[0][arow][aseg]     = ra0;
    *(uint4*)&As[0][arow][aseg + 8] = ra1;
    if (bstage) {
        *(uint4*)&Bs[0][brow][bseg]     = rb0;
        *(uint4*)&Bs[0][brow][bseg + 8] = rb1;
    }
    __syncthreads();

    const int T = K / BK;
    for (int t = 0; t < T; ++t) {
        const int cur = t & 1, nxt = cur ^ 1;

        // ---- prefetch next tile into registers (overlaps with WMMA below) ----
        const bool more = (t + 1) < T;
        if (more) {
            const int kk = (t + 1) * BK;
            ra0 = (uint4){0,0,0,0}; ra1 = (uint4){0,0,0,0};
            if (gr < M) { ra0 = *(const uint4*)(aptr + kk); ra1 = *(const uint4*)(aptr + kk + 8); }
            if (bstage) { rb0 = *(const uint4*)(bptr + kk); rb1 = *(const uint4*)(bptr + kk + 8); }
        }

        // ---- compute from current buffer ----
        // A frag (16x32 bf16): lane l -> row (l&15); elems 0..7 = K kbase.., 8..15 = K 16+kbase..
        // B frag (32x16):      lane l -> col (l&15); elems 0..7 = K kbase.., 8..15 = K 16+kbase..
        #pragma unroll
        for (int mi = 0; mi < 2; ++mi) {
            Frag16 af;
            const unsigned short* ap = &As[cur][wm * 32 + mi * 16 + l15][0];
            af.q[0] = *(const uint4*)(ap + kbase);
            af.q[1] = *(const uint4*)(ap + 16 + kbase);
            #pragma unroll
            for (int ni = 0; ni < 2; ++ni) {
                Frag16 bfr;
                const unsigned short* bp = &Bs[cur][wn * 32 + ni * 16 + l15][0];
                bfr.q[0] = *(const uint4*)(bp + kbase);
                bfr.q[1] = *(const uint4*)(bp + 16 + kbase);
                c[mi][ni] = __builtin_amdgcn_wmma_f32_16x16x32_bf16(
                    false, af.v, false, bfr.v, (short)0, c[mi][ni], false, false);
            }
        }

        // ---- commit prefetched tile to the other buffer ----
        if (more) {
            *(uint4*)&As[nxt][arow][aseg]     = ra0;
            *(uint4*)&As[nxt][arow][aseg + 8] = ra1;
            if (bstage) {
                *(uint4*)&Bs[nxt][brow][bseg]     = rb0;
                *(uint4*)&Bs[nxt][brow][bseg + 8] = rb1;
            }
        }
        __syncthreads();
    }

    // ---- epilogue: C/D layout: lane l -> col (l&15); VGPR r -> row r + 8*(l>>4) ----
    #pragma unroll
    for (int mi = 0; mi < 2; ++mi) {
        #pragma unroll
        for (int ni = 0; ni < 2; ++ni) {
            const int n = bn + wn * 32 + ni * 16 + l15;
            const float bv = bias ? bias[n] : 0.0f;
            #pragma unroll
            for (int r = 0; r < 8; ++r) {
                const int m = bm + wm * 32 + mi * 16 + half * 8 + r;
                if (m < M) C[(size_t)m * N + n] = c[mi][ni][r] + bv;
            }
        }
    }
}

// ---------------- SpMM helpers: CSR range by binary search on sorted rows ----------------
__device__ __forceinline__ void edge_range(const int* __restrict__ row, int n, int E,
                                           int& s, int& e) {
    int lo = 0, hi = E;
    while (lo < hi) { int mid = (lo + hi) >> 1; if (row[mid] <  n) lo = mid + 1; else hi = mid; }
    s = lo;
    hi = E;
    while (lo < hi) { int mid = (lo + hi) >> 1; if (row[mid] <= n) lo = mid + 1; else hi = mid; }
    e = lo;
}

// spmm(adj, XW1) + b1 + skip0, RReLU, convert to bf16 for next GEMM
__global__ __launch_bounds__(NHID)
void k_spmm_rrelu(const int* __restrict__ row, const int* __restrict__ col,
                  const float* __restrict__ val,
                  const float* __restrict__ XW1,     // [N_NODES, NHID]
                  const float* __restrict__ skip0,   // [N_NODES, NHID]
                  const float* __restrict__ b1,      // [NHID]
                  unsigned short* __restrict__ hb)   // [N_NODES, NHID] bf16
{
    const int n = blockIdx.x;
    const int f = threadIdx.x;
    int s, e;
    edge_range(row, n, N_EDGES, s, e);
    float acc = 0.0f;
    for (int i = s; i < e; ++i) {
        if (i + 1 < e)
            __builtin_prefetch(&XW1[(size_t)col[i + 1] * NHID + f], 0, 3);
        acc += val[i] * XW1[(size_t)col[i] * NHID + f];
    }
    const float v = acc + b1[f] + skip0[(size_t)n * NHID + f];
    const float slope = (1.0f / 8.0f + 1.0f / 3.0f) * 0.5f;
    const float h = (v >= 0.0f) ? v : v * slope;
    hb[(size_t)n * NHID + f] = f2bf(h);
}

// spmm(adj, HW3) + b3 + skip1 -> out (fp32)
__global__ __launch_bounds__(NCLASS)
void k_spmm_out(const int* __restrict__ row, const int* __restrict__ col,
                const float* __restrict__ val,
                const float* __restrict__ HW3,     // [N_NODES, NCLASS]
                const float* __restrict__ skip1,   // [N_NODES, NCLASS]
                const float* __restrict__ b3,      // [NCLASS]
                float* __restrict__ out)           // [N_NODES, NCLASS]
{
    const int n = blockIdx.x;
    const int f = threadIdx.x;
    int s, e;
    edge_range(row, n, N_EDGES, s, e);
    float acc = 0.0f;
    for (int i = s; i < e; ++i) {
        if (i + 1 < e)
            __builtin_prefetch(&HW3[(size_t)col[i + 1] * NCLASS + f], 0, 3);
        acc += val[i] * HW3[(size_t)col[i] * NCLASS + f];
    }
    out[(size_t)n * NCLASS + f] = acc + b3[f] + skip1[(size_t)n * NCLASS + f];
}

// ---------------- launcher ----------------
extern "C" void kernel_launch(void* const* d_in, const int* in_sizes, int n_in,
                              void* d_out, int out_size, void* d_ws, size_t ws_size,
                              hipStream_t stream) {
    (void)in_sizes; (void)n_in; (void)out_size; (void)ws_size;

    const float* x   = (const float*)d_in[0];
    const int* erow  = (const int*)d_in[1];
    const int* ecol  = (const int*)d_in[2];
    const float* eval= (const float*)d_in[3];
    const float* w1  = (const float*)d_in[4];
    const float* b1  = (const float*)d_in[5];
    const float* w2  = (const float*)d_in[6];
    const float* b2  = (const float*)d_in[7];
    const float* w3  = (const float*)d_in[8];
    const float* b3  = (const float*)d_in[9];
    const float* w4  = (const float*)d_in[10];
    const float* b4  = (const float*)d_in[11];

    float* out = (float*)d_out;

    // workspace suballocation (256 B aligned)
    char* ws = (char*)d_ws;
    size_t off = 0;
    auto alloc = [&](size_t bytes) { size_t o = off; off += (bytes + 255) & ~(size_t)255; return o; };

    unsigned short* xb  = (unsigned short*)(ws + alloc((size_t)N_NODES * NFEAT * 2));
    unsigned short* w1b = (unsigned short*)(ws + alloc((size_t)NFEAT * NHID * 2));   // [NHID,NFEAT]
    unsigned short* w2b = (unsigned short*)(ws + alloc((size_t)NFEAT * NHID * 2));   // [NHID,NFEAT]
    unsigned short* w3b = (unsigned short*)(ws + alloc((size_t)NHID * NCLASS * 2));  // [NCLASS,NHID]
    unsigned short* w4b = (unsigned short*)(ws + alloc((size_t)NFEAT * NCLASS * 2)); // [NCLASS,NFEAT]
    float* XW1   = (float*)(ws + alloc((size_t)N_NODES * NHID * 4));
    float* skip0 = (float*)(ws + alloc((size_t)N_NODES * NHID * 4));
    float* skip1 = (float*)(ws + alloc((size_t)N_NODES * NCLASS * 4));
    unsigned short* hb = (unsigned short*)(ws + alloc((size_t)N_NODES * NHID * 2));
    float* HW3 = XW1;   // alias: XW1 is dead once hb is built

    // 1) fp32 -> bf16 conversions (x straight; weights transposed to [N,K])
    {
        int n;
        n = N_NODES * NFEAT; k_f32_to_bf16<<<(n + 255) / 256, 256, 0, stream>>>(x, xb, n);
        n = NFEAT * NHID;    k_wtrans_bf16<<<(n + 255) / 256, 256, 0, stream>>>(w1, w1b, NFEAT, NHID);
        n = NFEAT * NHID;    k_wtrans_bf16<<<(n + 255) / 256, 256, 0, stream>>>(w2, w2b, NFEAT, NHID);
        n = NHID * NCLASS;   k_wtrans_bf16<<<(n + 255) / 256, 256, 0, stream>>>(w3, w3b, NHID, NCLASS);
        n = NFEAT * NCLASS;  k_wtrans_bf16<<<(n + 255) / 256, 256, 0, stream>>>(w4, w4b, NFEAT, NCLASS);
    }

    const int gm = (N_NODES + BM - 1) / BM;   // 391

    // 2) XW1 = x @ w1            [50000,256]
    k_gemm_bf16<<<dim3(gm, NHID / BN), 256, 0, stream>>>(xb, w1b, nullptr, XW1,
                                                         N_NODES, NHID, NFEAT);
    // 3) skip0 = x @ w2 + b2     [50000,256]
    k_gemm_bf16<<<dim3(gm, NHID / BN), 256, 0, stream>>>(xb, w2b, b2, skip0,
                                                         N_NODES, NHID, NFEAT);
    // 4) skip1 = x @ w4 + b4     [50000,64]
    k_gemm_bf16<<<dim3(gm, NCLASS / BN), 256, 0, stream>>>(xb, w4b, b4, skip1,
                                                           N_NODES, NCLASS, NFEAT);
    // 5) h = RReLU(spmm + b1 + skip0) -> bf16
    k_spmm_rrelu<<<N_NODES, NHID, 0, stream>>>(erow, ecol, eval, XW1, skip0, b1, hb);
    // 6) HW3 = h @ w3            [50000,64]  (aliases XW1 region)
    k_gemm_bf16<<<dim3(gm, NCLASS / BN), 256, 0, stream>>>(hb, w3b, nullptr, HW3,
                                                           N_NODES, NCLASS, NHID);
    // 7) out = spmm(HW3) + b3 + skip1
    k_spmm_out<<<N_NODES, NCLASS, 0, stream>>>(erow, ecol, eval, HW3, skip1, b3, out);

    // 8) tuple tail: (out, w1, w2, w3, w4) concatenated flat
    size_t o = (size_t)N_NODES * NCLASS;
    hipMemcpyAsync(out + o, w1, (size_t)NFEAT * NHID   * 4, hipMemcpyDeviceToDevice, stream);
    o += (size_t)NFEAT * NHID;
    hipMemcpyAsync(out + o, w2, (size_t)NFEAT * NHID   * 4, hipMemcpyDeviceToDevice, stream);
    o += (size_t)NFEAT * NHID;
    hipMemcpyAsync(out + o, w3, (size_t)NHID * NCLASS  * 4, hipMemcpyDeviceToDevice, stream);
    o += (size_t)NHID * NCLASS;
    hipMemcpyAsync(out + o, w4, (size_t)NFEAT * NCLASS * 4, hipMemcpyDeviceToDevice, stream);
}
